// mlpNetwork_89438398972151
// MI455X (gfx1250) — compile-verified, matrix operand order checked by reading, not run on hardware
//
#include <hip/hip_runtime.h>

// ---------------------------------------------------------------------------
// CDNA5 (gfx1250) implementation of the 3-layer CConv graph network.
// - All heavy GEMMs use v_wmma_f32_16x16x32_f16 (fp16 in, f32 accumulate).
// - The 64->256 output linear is fused with the per-edge matvec + scatter so
//   the [E,256] filter tensor (205MB/layer) is never materialized in HBM.
// - First hidden layer (scalar input) BatchNorm is computed analytically from
//   attr statistics, removing one full [E,64] pass.
// - Tile staging uses GLOBAL_LOAD_ASYNC_TO_LDS_B128 (ASYNCcnt-tracked direct
//   memory->LDS path) when the toolchain exposes the builtin.
// - Working set (Y 51.2MB f32 + X 25.6MB fp16 + h bufs) fits the 192MB L2,
//   so default temporal hints keep all inter-kernel traffic on-chip.
// ---------------------------------------------------------------------------

typedef __attribute__((ext_vector_type(16))) _Float16 v16h;
typedef __attribute__((ext_vector_type(4)))  _Float16 v4h;
typedef __attribute__((ext_vector_type(8)))  float    v8f;
typedef __attribute__((ext_vector_type(4)))  int      v4i;

#define NVERT  25000
#define NEDGE  200000
#define FDIM   16
#define HDIM   64
#define ODIM   256        // FDIM*FDIM
#define BN_EPS 1e-5f

#ifndef __has_builtin
#define __has_builtin(x) 0
#endif
#if __has_builtin(__builtin_amdgcn_global_load_async_to_lds_b128)
#define HAVE_ASYNC_LDS 1
#else
#define HAVE_ASYNC_LDS 0
#endif

// 16B memory -> LDS copy; async (ASYNCcnt) when available.
// Builtin params are address-space-qualified v4i pointers.
__device__ __forceinline__ void copy16_to_lds(const void* g, void* l) {
#if HAVE_ASYNC_LDS
  __builtin_amdgcn_global_load_async_to_lds_b128(
      (__attribute__((address_space(1))) v4i*)g,
      (__attribute__((address_space(3))) v4i*)l, 0, 0);
#else
  *(uint4*)l = *(const uint4*)g;
#endif
}

__device__ __forceinline__ void async_drain() {
#if HAVE_ASYNC_LDS
#if __has_builtin(__builtin_amdgcn_s_wait_asynccnt)
  __builtin_amdgcn_s_wait_asynccnt(0);
#else
  asm volatile("s_wait_asynccnt 0x0" ::: "memory");
#endif
#endif
}

// K-index mapping for the 16-bit A/B WMMA fragment layout (ISA 7.12.2):
// lanes 0-15 hold K = {e} for e<8 and K = {e+8} for e>=8;
// lanes 16-31 hold the same pattern shifted by +8.
__device__ __forceinline__ int kmap(int e, int hi) {
  return e + (hi ? 8 : 0) + ((e >= 8) ? 8 : 0);
}

// ---------------------------------------------------------------------------
__global__ void k_zero(float* p, int n) {
  int i = blockIdx.x * blockDim.x + threadIdx.x;
  int stride = gridDim.x * blockDim.x;
  for (; i < n; i += stride) p[i] = 0.f;
}

// Scalar mean/var of edge_attr (single reduction, reused by all 3 layers).
__global__ __launch_bounds__(256)
void k_attr_stats(const float* __restrict__ attr, float* __restrict__ st) {
  __shared__ float s1[256], s2[256];
  int tid = threadIdx.x;
  float a1 = 0.f, a2 = 0.f;
  for (int i = blockIdx.x * 256 + tid; i < NEDGE; i += gridDim.x * 256) {
    float a = attr[i];
    a1 += a; a2 += a * a;
  }
  s1[tid] = a1; s2[tid] = a2;
  __syncthreads();
  for (int off = 128; off > 0; off >>= 1) {
    if (tid < off) { s1[tid] += s1[tid + off]; s2[tid] += s2[tid + off]; }
    __syncthreads();
  }
  if (tid == 0) { atomicAdd(&st[0], s1[0]); atomicAdd(&st[1], s2[0]); }
}

// Per-layer prep: analytic BN folding for hidden layer 1 (scalar input),
// fp16 conversion of W2/W3/Wo, zero the running column stats.
__global__ __launch_bounds__(256)
void k_prep(const float* __restrict__ W1, const float* __restrict__ b1,
            const float* __restrict__ g1, const float* __restrict__ be1,
            const float* __restrict__ W2, const float* __restrict__ W3,
            const float* __restrict__ Wo, const float* __restrict__ st,
            float* __restrict__ alpha, float* __restrict__ gammao,
            float* __restrict__ hsum, float* __restrict__ hsq,
            _Float16* __restrict__ W2h, _Float16* __restrict__ W3h,
            _Float16* __restrict__ Woh) {
  int tid = threadIdx.x;
  if (tid < HDIM) {
    float mean_a = st[0] * (1.f / NEDGE);
    float var_a  = st[1] * (1.f / NEDGE) - mean_a * mean_a;
    float w1 = W1[tid];                       // W1 is [64,1]
    float rstd = rsqrtf(w1 * w1 * var_a + BN_EPS);
    float al = g1[tid] * w1 * rstd;
    alpha[tid]  = al;
    gammao[tid] = be1[tid] - al * mean_a;     // b1 cancels against mu
    (void)b1;
    hsum[tid] = 0.f; hsq[tid] = 0.f;
  }
  for (int i = tid; i < HDIM * HDIM; i += 256) {
    W2h[i] = (_Float16)W2[i];
    W3h[i] = (_Float16)W3[i];
  }
  for (int i = tid; i < ODIM * HDIM; i += 256) Woh[i] = (_Float16)Wo[i];
}

// Hidden-layer GEMM: Y[E,64] = X[E,64] @ W^T + b, plus per-column batch stats.
// modeA==1: X is generated on the fly as relu(alpha*attr+gamma) (layer-1 fold).
// Block = 128 threads = 4 waves; each wave owns one 16-row M-tile, 4 N-tiles,
// K=64 as two 16x16x32 fp16 WMMA steps.
__global__ __launch_bounds__(128)
void k_gemm(const float* __restrict__ attr, const _Float16* __restrict__ Xin,
            int modeA,
            const float* __restrict__ alpha, const float* __restrict__ gammao,
            const _Float16* __restrict__ Wh, const float* __restrict__ bias,
            float* __restrict__ Y,
            float* __restrict__ hsum, float* __restrict__ hsq) {
  __shared__ __align__(16) _Float16 Wt[HDIM * HDIM];   // 8KB
  __shared__ __align__(16) _Float16 Xt[64 * HDIM];     // 8KB
  __shared__ float csum[HDIM], csq[HDIM];

  int tid  = threadIdx.x;
  int wave = tid >> 5;
  int lane = tid & 31;
  int e0   = blockIdx.x * 64;

  for (int i = tid; i < (HDIM * HDIM) / 8; i += 128)
    copy16_to_lds((const uint4*)Wh + i, (uint4*)Wt + i);
  if (tid < HDIM) { csum[tid] = 0.f; csq[tid] = 0.f; }

  if (modeA) {
    for (int i = tid; i < 64 * HDIM; i += 128) {
      int m = i >> 6, j = i & 63;
      float v = alpha[j] * attr[e0 + m] + gammao[j];
      Xt[i] = (_Float16)fmaxf(v, 0.f);
    }
  } else {
    const uint4* src = (const uint4*)(Xin + (size_t)e0 * HDIM);
    for (int i = tid; i < (64 * HDIM) / 8; i += 128)
      copy16_to_lds(src + i, (uint4*)Xt + i);
  }
  async_drain();
  __syncthreads();

  int lrow = lane & 15;
  int hi   = lane >> 4;

  v8f c[4];
#pragma unroll
  for (int n = 0; n < 4; n++) c[n] = {};

#pragma unroll
  for (int s = 0; s < 2; s++) {
    v16h a;
#pragma unroll
    for (int e = 0; e < 16; e++)
      a[e] = Xt[(wave * 16 + lrow) * HDIM + s * 32 + kmap(e, hi)];
#pragma unroll
    for (int n = 0; n < 4; n++) {
      v16h b;
#pragma unroll
      for (int e = 0; e < 16; e++)
        b[e] = Wt[(n * 16 + lrow) * HDIM + s * 32 + kmap(e, hi)];
      c[n] = __builtin_amdgcn_wmma_f32_16x16x32_f16(
          false, a, false, b, (short)0, c[n], false, false);
    }
  }

#pragma unroll
  for (int n = 0; n < 4; n++) {
    int j = n * 16 + lrow;
    float bj = bias[j];
    float s1 = 0.f, s2 = 0.f;
#pragma unroll
    for (int r = 0; r < 8; r++) {
      float v = c[n][r] + bj;
      int m = wave * 16 + r + hi * 8;
      Y[(size_t)(e0 + m) * HDIM + j] = v;
      s1 += v; s2 += v * v;
    }
    atomicAdd(&csum[j], s1);
    atomicAdd(&csq[j], s2);
  }
  __syncthreads();
  if (tid < HDIM) {
    atomicAdd(&hsum[tid], csum[tid]);
    atomicAdd(&hsq[tid],  csq[tid]);
  }
}

// Finalize BN: fold (gamma, beta, mu, rstd) into per-column scale/shift.
__global__ void k_bnfin(const float* __restrict__ g, const float* __restrict__ be,
                        float* __restrict__ hsum, float* __restrict__ hsq,
                        float* __restrict__ scale, float* __restrict__ shift) {
  int j = threadIdx.x;
  if (j < HDIM) {
    float mu  = hsum[j] * (1.f / NEDGE);
    float var = hsq[j] * (1.f / NEDGE) - mu * mu;
    float sc  = g[j] * rsqrtf(var + BN_EPS);
    scale[j] = sc;
    shift[j] = be[j] - mu * sc;
    hsum[j] = 0.f; hsq[j] = 0.f;   // ready for the next GEMM's stats
  }
}

// Apply BN + ReLU, emit fp16 activations for the next WMMA stage.
__global__ __launch_bounds__(256)
void k_bnapply(const float* __restrict__ Y, _Float16* __restrict__ X,
               const float* __restrict__ scale, const float* __restrict__ shift) {
  int i = blockIdx.x * blockDim.x + threadIdx.x;
  int stride = gridDim.x * blockDim.x;
  const int n4 = (NEDGE * HDIM) / 4;
  for (; i < n4; i += stride) {
    float4 y = ((const float4*)Y)[i];
    int j = (i * 4) & 63;
    v4h o;
    o[0] = (_Float16)fmaxf(y.x * scale[j + 0] + shift[j + 0], 0.f);
    o[1] = (_Float16)fmaxf(y.y * scale[j + 1] + shift[j + 1], 0.f);
    o[2] = (_Float16)fmaxf(y.z * scale[j + 2] + shift[j + 2], 0.f);
    o[3] = (_Float16)fmaxf(y.w * scale[j + 3] + shift[j + 3], 0.f);
    *(v4h*)(X + (size_t)i * 4) = o;
  }
}

// Fused output stage: per 16-edge tile, w = X3 @ Wo^T + bo via WMMA
// (16 N-tiles, K=64), immediately contracted with gathered h[src] and
// scattered to h[dst] with f32 atomics. w never hits global memory.
__global__ __launch_bounds__(128)
void k_msg(const _Float16* __restrict__ X3, const _Float16* __restrict__ Woh,
           const float* __restrict__ bo, const float* __restrict__ hcur,
           const int* __restrict__ esrc, const int* __restrict__ edst,
           float* __restrict__ hnext, int relu_gather) {
  __shared__ __align__(16) _Float16 Wl[ODIM * HDIM];       // 32KB, block-shared
  __shared__ __align__(16) _Float16 X3t[4][16 * HDIM];     // 8KB, per-wave tile
  __shared__ __align__(16) float    hs[4][16 * FDIM];      // 4KB, gathered h[src]

  int tid  = threadIdx.x;
  int wave = tid >> 5;
  int lane = tid & 31;
  int e0   = (blockIdx.x * 4 + wave) * 16;

  __builtin_prefetch(esrc + e0, 0, 0);     // global_prefetch_b8
  __builtin_prefetch(edst + e0, 0, 0);

  for (int i = tid; i < (ODIM * HDIM) / 8; i += 128)
    copy16_to_lds((const uint4*)Woh + i, (uint4*)Wl + i);

  {
    const uint4* s = (const uint4*)(X3 + (size_t)e0 * HDIM);
    uint4* d = (uint4*)X3t[wave];
    for (int i = lane; i < (16 * HDIM) / 8; i += 32)
      copy16_to_lds(s + i, d + i);
  }
  for (int t = lane; t < 64; t += 32) {          // 16 rows x 4 float4
    int m = t >> 2, p = t & 3;
    int sv = esrc[e0 + m];
    float4 v = ((const float4*)(hcur + (size_t)sv * FDIM))[p];
    if (relu_gather) {
      v.x = fmaxf(v.x, 0.f); v.y = fmaxf(v.y, 0.f);
      v.z = fmaxf(v.z, 0.f); v.w = fmaxf(v.w, 0.f);
    }
    ((float4*)&hs[wave][m * FDIM])[p] = v;
  }
  async_drain();
  __syncthreads();

  int col = lane & 15;
  int hi  = lane >> 4;

  v16h a[2];
#pragma unroll
  for (int s = 0; s < 2; s++)
#pragma unroll
    for (int e = 0; e < 16; e++)
      a[s][e] = X3t[wave][col * HDIM + s * 32 + kmap(e, hi)];

  float msg[8];
#pragma unroll
  for (int r = 0; r < 8; r++) msg[r] = 0.f;

#pragma unroll
  for (int i = 0; i < FDIM; i++) {               // input-channel tile = N-tile i
    v8f c = {};
#pragma unroll
    for (int s = 0; s < 2; s++) {
      v16h b;
#pragma unroll
      for (int e = 0; e < 16; e++)
        b[e] = Wl[(i * 16 + col) * HDIM + s * 32 + kmap(e, hi)];
      c = __builtin_amdgcn_wmma_f32_16x16x32_f16(
          false, a[s], false, b, (short)0, c, false, false);
    }
    float bj = bo[i * 16 + col];
#pragma unroll
    for (int r = 0; r < 8; r++) {
      int m = r + hi * 8;
      msg[r] += (c[r] + bj) * hs[wave][m * FDIM + i];
    }
  }

#pragma unroll
  for (int r = 0; r < 8; r++) {
    int m = r + hi * 8;
    int dv = edst[e0 + m];
    atomicAdd(&hnext[(size_t)dv * FDIM + col], msg[r]);
  }
}

// ---------------------------------------------------------------------------
extern "C" void kernel_launch(void* const* d_in, const int* in_sizes, int n_in,
                              void* d_out, int out_size, void* d_ws, size_t ws_size,
                              hipStream_t stream) {
  const float* x    = (const float*)d_in[0];
  const float* attr = (const float*)d_in[1];
  const int*   esrc = (const int*)d_in[2];
  const int*   edst = (const int*)d_in[3];
  // edge_params pytree order: per layer, hidden[(W,b,g,be)x3] then out(Wo,bo).
  auto P = [&](int l, int k) -> const float* {
    return (const float*)d_in[4 + l * 14 + k];
  };

  char* w = (char*)d_ws;
  float*    Y     = (float*)w;     w += (size_t)NEDGE * HDIM * 4;   // 51.2 MB
  _Float16* X     = (_Float16*)w;  w += (size_t)NEDGE * HDIM * 2;   // 25.6 MB
  float*    hbuf0 = (float*)w;     w += (size_t)NVERT * FDIM * 4;   // 1.6 MB
  float*    hbuf1 = (float*)w;     w += (size_t)NVERT * FDIM * 4;   // 1.6 MB
  float*    st    = (float*)w;     w += 256;                        // attr stats
  float*    hsum  = (float*)w;     w += HDIM * 4;
  float*    hsq   = (float*)w;     w += HDIM * 4;
  float*    alpha = (float*)w;     w += HDIM * 4;
  float*    gamm  = (float*)w;     w += HDIM * 4;
  float*    scale = (float*)w;     w += HDIM * 4;
  float*    shift = (float*)w;     w += HDIM * 4;
  _Float16* W2h   = (_Float16*)w;  w += HDIM * HDIM * 2;
  _Float16* W3h   = (_Float16*)w;  w += HDIM * HDIM * 2;
  _Float16* Woh   = (_Float16*)w;  w += ODIM * HDIM * 2;

  float* out = (float*)d_out;

  const int gemm_blocks = NEDGE / 64;   // 3125
  const int msg_blocks  = NEDGE / 64;   // 4 tiles of 16 edges per block

  k_zero<<<1, 64, 0, stream>>>(st, 2);
  k_attr_stats<<<256, 256, 0, stream>>>(attr, st);

  for (int l = 0; l < 3; l++) {
    const float* hc = (l == 0) ? x : ((l == 1) ? hbuf0 : hbuf1);
    float*       hn = (l == 0) ? hbuf0 : ((l == 1) ? hbuf1 : out);

    k_prep<<<1, 256, 0, stream>>>(P(l,0), P(l,1), P(l,2), P(l,3),
                                  P(l,4), P(l,8), P(l,12), st,
                                  alpha, gamm, hsum, hsq, W2h, W3h, Woh);

    // hidden layer 2 (X1 generated from attr on the fly)
    k_gemm<<<gemm_blocks, 128, 0, stream>>>(attr, X, 1, alpha, gamm,
                                            W2h, P(l,5), Y, hsum, hsq);
    k_bnfin<<<1, 64, 0, stream>>>(P(l,6), P(l,7), hsum, hsq, scale, shift);
    k_bnapply<<<2048, 256, 0, stream>>>(Y, X, scale, shift);

    // hidden layer 3
    k_gemm<<<gemm_blocks, 128, 0, stream>>>(attr, X, 0, alpha, gamm,
                                            W3h, P(l,9), Y, hsum, hsq);
    k_bnfin<<<1, 64, 0, stream>>>(P(l,10), P(l,11), hsum, hsq, scale, shift);
    k_bnapply<<<2048, 256, 0, stream>>>(Y, X, scale, shift);

    // fused output linear + gather-matvec + scatter
    k_zero<<<512, 256, 0, stream>>>(hn, NVERT * FDIM);
    k_msg<<<msg_blocks, 128, 0, stream>>>(X, Woh, P(l,13), hc, esrc, edst,
                                          hn, (l >= 1) ? 1 : 0);
  }
}
